// Deformable_conv_25615184953697
// MI455X (gfx1250) — matrix operand order repaired and checked by reading, not compiled, not run on hardware
//
#include <hip/hip_runtime.h>
#include <math.h>

typedef __attribute__((ext_vector_type(16))) _Float16 v16h;
typedef __attribute__((ext_vector_type(8)))  _Float16 v8h;
typedef __attribute__((ext_vector_type(8)))  float    v8f;
typedef __attribute__((ext_vector_type(4)))  float    v4f;

// ---- problem constants (from reference) ----
constexpr int B  = 8,  C  = 64, H = 128, W = 128;
constexpr int Ho = 64, Wo = 128;
constexpr int O  = 64, K2 = 9;
constexpr int KG = C * K2;            // 576 = GEMM K dimension
constexpr int HW = Ho * Wo;           // 8192
constexpr int NPIX = B * HW;          // 65536 = GEMM N dimension

// ---------------------------------------------------------------------------
// Kernel 1: offset (18ch) + mask (9ch) 3x3 conv, stride (2,1), pad (1,1).
// One block per (b, ho); one thread per wo. Weights staged in LDS (27->28 pad).
// ---------------------------------------------------------------------------
__global__ __launch_bounds__(128) void offmask_kernel(
    const float* __restrict__ X,     const float* __restrict__ Woff,
    const float* __restrict__ Boff,  const float* __restrict__ Wmask,
    const float* __restrict__ Bmask, float* __restrict__ OFF,
    float* __restrict__ MASK)
{
  __shared__ alignas(16) float wt[576 * 28];   // [c*9 + ky*3 + kx][o], o padded to 28
  const int tid = threadIdx.x;
  for (int j = tid; j < 576 * 28; j += 128) {
    const int o = j % 28, pos = j / 28;
    float v = 0.0f;
    if (o < 18)      v = Woff[o * 576 + pos];
    else if (o < 27) v = Wmask[(o - 18) * 576 + pos];
    wt[j] = v;
  }
  __syncthreads();

  const int bid = blockIdx.x;
  const int b = bid >> 6, ho = bid & 63;
  const int wo = tid;

  float acc[28];
#pragma unroll
  for (int o = 0; o < 28; ++o) acc[o] = 0.0f;
#pragma unroll
  for (int o = 0; o < 18; ++o) acc[o] = Boff[o];
#pragma unroll
  for (int o = 0; o < 9;  ++o) acc[18 + o] = Bmask[o];

  for (int c = 0; c < C; ++c) {
#pragma unroll
    for (int r = 0; r < 3; ++r) {
      const int hi = 2 * ho - 1 + r;
      float xm = 0.0f, x0 = 0.0f, xp = 0.0f;
      if (hi >= 0 && hi < H) {
        const float* xr = X + (((size_t)(b * C + c) * H + hi) * W);
        x0 = xr[wo];
        if (wo > 0)     xm = xr[wo - 1];
        if (wo < W - 1) xp = xr[wo + 1];
      }
      const float xv3[3] = { xm, x0, xp };
#pragma unroll
      for (int kx = 0; kx < 3; ++kx) {
        const float xv = xv3[kx];
        const v4f* wr = reinterpret_cast<const v4f*>(&wt[(c * 9 + r * 3 + kx) * 28]);
#pragma unroll
        for (int q = 0; q < 7; ++q) {
          const v4f w4 = wr[q];
#pragma unroll
          for (int e = 0; e < 4; ++e) acc[q * 4 + e] += w4[e] * xv;
        }
      }
    }
  }

  const size_t sp = (size_t)ho * Wo + wo;
#pragma unroll
  for (int o = 0; o < 18; ++o)
    OFF[(size_t)(b * 18 + o) * HW + sp] = acc[o];
#pragma unroll
  for (int o = 0; o < 9; ++o)
    MASK[(size_t)(b * 9 + o) * HW + sp] = 1.0f / (1.0f + expf(-acc[18 + o]));
}

// ---------------------------------------------------------------------------
// Kernel 1b: w_conv fp32 -> f16. Flat (o,c,ky,kx) order == A[m][kg] row-major.
// ---------------------------------------------------------------------------
__global__ __launch_bounds__(256) void w_to_f16_kernel(
    const float* __restrict__ w, _Float16* __restrict__ out, int n)
{
  const int i = blockIdx.x * 256 + threadIdx.x;
  if (i < n) out[i] = (_Float16)w[i];
}

// ---------------------------------------------------------------------------
// Kernel 2: fused bilinear sampling + WMMA GEMM.
// Block = 256 threads (8 waves) owns 32 consecutive output pixels (one row seg).
//   Phase 0: per-(pixel,tap) sample coords + mask -> LDS
//   Phase 1: 576x32 f16 B-tile via bilinear gather -> LDS
//   Phase 2: D(64x32) = A(64x576) x B(576x32) with v_wmma_f32_16x16x32_f16
// ---------------------------------------------------------------------------
constexpr int TN = 32;
constexpr int BSTRIDE = 584;   // 576 + 8: 16B-aligned rows, bank-friendly

__global__ __launch_bounds__(256) void dcn_main_kernel(
    const float* __restrict__ X,    const float* __restrict__ OFF,
    const float* __restrict__ MASK, const _Float16* __restrict__ Ah,
    float* __restrict__ OUT)
{
  __shared__ alignas(16) _Float16 Bs[TN * BSTRIDE];  // [n][kg]
  __shared__ float py_s[TN * K2];
  __shared__ float px_s[TN * K2];
  __shared__ float m_s [TN * K2];

  const int tid  = threadIdx.x;
  const int pix0 = blockIdx.x * TN;
  const int b    = pix0 >> 13;          // / (Ho*Wo)
  const int ho   = (pix0 >> 7) & 63;
  const int wo0  = pix0 & 127;          // tile never crosses a row (32 | 128)

  // ---- phase 0: sampling coordinates -------------------------------------
  for (int i = tid; i < TN * K2; i += 256) {
    const int n = i / K2, k2 = i % K2;
    const int wo = wo0 + n;
    const size_t sp = (size_t)ho * Wo + wo;
    const float offy = OFF[(size_t)(b * 18 + 2 * k2 + 0) * HW + sp];
    const float offx = OFF[(size_t)(b * 18 + 2 * k2 + 1) * HW + sp];
    py_s[i] = offy + (float)(k2 / 3) + (float)(2 * ho - 1);
    px_s[i] = offx + (float)(k2 % 3) + (float)(wo - 1);
    m_s[i]  = MASK[(size_t)(b * 9 + k2) * HW + sp];
  }
  __syncthreads();

  // ---- phase 1: bilinear gather -> f16 B tile ----------------------------
  for (int idx = tid; idx < TN * KG; idx += 256) {
    const int n  = idx & (TN - 1);      // consecutive lanes -> consecutive wo
    const int kg = idx >> 5;
    const int c  = kg / K2;
    const int k2 = kg - c * K2;
    const int aux = n * K2 + k2;
    const float py = py_s[aux], px = px_s[aux], mv = m_s[aux];
    const float y0f = floorf(py), x0f = floorf(px);
    const float dy = py - y0f,  dx = px - x0f;
    const int iy = (int)y0f, ix = (int)x0f;
    const float* xc = X + ((size_t)(b * C + c) << 14);   // 128*128 plane
    const bool y0ok = (iy >= 0)     & (iy < H);
    const bool y1ok = (iy + 1 >= 0) & (iy + 1 < H);
    const bool x0ok = (ix >= 0)     & (ix < W);
    const bool x1ok = (ix + 1 >= 0) & (ix + 1 < W);
    float v00 = 0.f, v01 = 0.f, v10 = 0.f, v11 = 0.f;
    if (y0ok & x0ok) v00 = xc[iy * W + ix];
    if (y0ok & x1ok) v01 = xc[iy * W + ix + 1];
    if (y1ok & x0ok) v10 = xc[(iy + 1) * W + ix];
    if (y1ok & x1ok) v11 = xc[(iy + 1) * W + ix + 1];
    const float v = (1.f - dy) * ((1.f - dx) * v00 + dx * v01)
                  +        dy  * ((1.f - dx) * v10 + dx * v11);
    Bs[n * BSTRIDE + kg] = (_Float16)(v * mv);
  }
  __syncthreads();

  // ---- phase 2: WMMA GEMM (all lanes active, uniform flow) ---------------
  const int lane = tid & 31;
  const int wv   = tid >> 5;
  const int m0 = (wv >> 1) * 16;        // 4 M-tiles
  const int n0 = (wv & 1) * 16;         // 2 N-tiles
  // A fragment lanes: M = m0 + lane%16 ; lanes<16 hold K {0-7,16-23}, else {8-15,24-31}
  const int mA = m0 + (lane & 15);
  const int kA = (lane < 16) ? 0 : 8;
  // B fragment lanes: N = n0 + lane%16 ; lanes<16 hold K {0-7,16-23}, else {8-15,24-31}
  const int nB = n0 + (lane & 15);
  const int kB = (lane < 16) ? 0 : 8;
  const _Float16* aRow = Ah + (size_t)mA * KG;
  const _Float16* bRow = &Bs[nB * BSTRIDE];

  v8f acc = {};
  for (int kk = 0; kk < KG; kk += 32) {
    const v8h alo = *reinterpret_cast<const v8h*>(aRow + kk + kA);
    const v8h ahi = *reinterpret_cast<const v8h*>(aRow + kk + 16 + kA);
    const v8h blo = *reinterpret_cast<const v8h*>(bRow + kk + kB);
    const v8h bhi = *reinterpret_cast<const v8h*>(bRow + kk + 16 + kB);
    const v16h a = __builtin_shufflevector(alo, ahi,
        0,1,2,3,4,5,6,7,8,9,10,11,12,13,14,15);
    const v16h bm = __builtin_shufflevector(blo, bhi,
        0,1,2,3,4,5,6,7,8,9,10,11,12,13,14,15);
    acc = __builtin_amdgcn_wmma_f32_16x16x32_f16(
        /*neg_a=*/false, a, /*neg_b=*/false, bm,
        /*c_mod=*/(short)0, acc, /*reuse_a=*/false, /*reuse_b=*/false);
  }

  // C/D layout: VGPR r -> M = r (lanes 0-15) or r+8 (lanes 16-31); lane%16 -> N
  const int mlane = m0 + ((lane >> 4) << 3);
  const int nlane = n0 + (lane & 15);
  const int wo = wo0 + nlane;
#pragma unroll
  for (int r = 0; r < 8; ++r) {
    const int m = mlane + r;
    OUT[((size_t)(b * O + m) * Ho + ho) * Wo + wo] = acc[r];
  }
}

// ---------------------------------------------------------------------------
extern "C" void kernel_launch(void* const* d_in, const int* in_sizes, int n_in,
                              void* d_out, int out_size, void* d_ws, size_t ws_size,
                              hipStream_t stream)
{
  const float* X     = (const float*)d_in[0];
  const float* Woff  = (const float*)d_in[1];
  const float* Boff  = (const float*)d_in[2];
  const float* Wmask = (const float*)d_in[3];
  const float* Bmask = (const float*)d_in[4];
  const float* Wconv = (const float*)d_in[5];
  float* OUT = (float*)d_out;

  // workspace layout: OFF (4.72 MB) | MASK (2.36 MB) | Ah f16 (72 KB)
  float*     OFF  = (float*)d_ws;
  float*     MASK = OFF + (size_t)B * 18 * HW;
  _Float16*  Ah   = (_Float16*)(MASK + (size_t)B * 9 * HW);

  offmask_kernel<<<B * Ho, 128, 0, stream>>>(X, Woff, Boff, Wmask, Bmask, OFF, MASK);
  w_to_f16_kernel<<<(O * KG + 255) / 256, 256, 0, stream>>>(Wconv, Ah, O * KG);
  dcn_main_kernel<<<NPIX / TN, 256, 0, stream>>>(X, OFF, MASK, Ah, OUT);
}
